// PrototypeMemory_11897059410793
// MI455X (gfx1250) — matrix operand order repaired and checked by reading, not compiled, not run on hardware
//
#include <hip/hip_runtime.h>
#include <hip/hip_bf16.h>
#include <math.h>

// ---------------------------------------------------------------------------
// PrototypeMemory: prototype EMA update + (-euclidean distance) logits.
// Strategy: fp32 normalization -> f16 unit vectors -> v_wmma_f32_16x16x32_f16
// distance GEMM (fp32 accumulate). Output write (524MB) is the roofline floor.
// ---------------------------------------------------------------------------

#define NROWS   131072
#define FD      256
#define NCLS    1000
#define CPAD    1024      // pad classes to 64-column groups (16 groups)
#define CGROUPS 16        // CPAD / 64
#define L2EPS   1e-12f

typedef __attribute__((ext_vector_type(16))) _Float16 v16h;
typedef __attribute__((ext_vector_type(8)))  _Float16 v8h;
typedef __attribute__((ext_vector_type(8)))  float    v8f;
typedef __attribute__((ext_vector_type(4)))  float    v4f;

__device__ __forceinline__ float wave_sum(float v) {
#pragma unroll
  for (int m = 16; m >= 1; m >>= 1) v += __shfl_xor(v, m, 32);
  return v;
}

__device__ __forceinline__ void atomic_add_f32(float* p, float v) {
  __hip_atomic_fetch_add(p, v, __ATOMIC_RELAXED, __HIP_MEMORY_SCOPE_AGENT);
}

// ---------------------------------------------------------------------------
// zero scratch (sums + counts)
// ---------------------------------------------------------------------------
__global__ void k_zero(float* __restrict__ p, int n) {
  int i = blockIdx.x * blockDim.x + threadIdx.x;
  if (i < n) p[i] = 0.0f;
}

// ---------------------------------------------------------------------------
// Normalize each feature row (fp32), store f16 copy + sum(fn*fn), and do the
// per-class segment sums/counts with native f32 global atomics.
// One wave per row; 8 floats per lane.
// ---------------------------------------------------------------------------
__global__ __launch_bounds__(256) void k_norm_accum(
    const float* __restrict__ feat, const int* __restrict__ labels,
    _Float16* __restrict__ fn_h, float* __restrict__ fsq,
    float* __restrict__ sums, float* __restrict__ counts)
{
  const int gid  = blockIdx.x * blockDim.x + threadIdx.x;
  const int row  = gid >> 5;
  const int lane = gid & 31;
  if (row >= NROWS) return;

  const float* fr = feat + (size_t)row * FD + lane * 8;
  v4f x0 = *(const v4f*)fr;
  v4f x1 = *(const v4f*)(fr + 4);
  float x[8];
#pragma unroll
  for (int j = 0; j < 4; ++j) { x[j] = x0[j]; x[4 + j] = x1[j]; }

  float ss = 0.0f;
#pragma unroll
  for (int j = 0; j < 8; ++j) ss += x[j] * x[j];
  ss = wave_sum(ss);
  const float inv = 1.0f / fmaxf(sqrtf(ss), L2EPS);

  v8h h;
  float fn[8];
#pragma unroll
  for (int j = 0; j < 8; ++j) { fn[j] = x[j] * inv; h[j] = (_Float16)fn[j]; }
  *(v8h*)(fn_h + (size_t)row * FD + lane * 8) = h;

  const int lab = labels[row];
  float* srow = sums + (size_t)lab * FD + lane * 8;
#pragma unroll
  for (int j = 0; j < 8; ++j) atomic_add_f32(srow + j, fn[j]);
  if (lane == 0) {
    atomic_add_f32(counts + lab, 1.0f);
    fsq[row] = ss * inv * inv;   // == sum(fn*fn), ~1.0
  }
}

// ---------------------------------------------------------------------------
// Prototype update: class_mean -> l2norm -> EMA -> select -> l2norm.
// One wave per (padded) class. Pads write zeros.
// ---------------------------------------------------------------------------
__global__ __launch_bounds__(32) void k_proto(
    const float* __restrict__ protos,
    const unsigned char* __restrict__ initialized,   // numpy bool: 1 byte/elem
    const int* __restrict__ epoch_p,
    const float* __restrict__ sums, const float* __restrict__ counts,
    _Float16* __restrict__ pn_h, float* __restrict__ pn_sq)
{
  const int c    = blockIdx.x;
  const int lane = threadIdx.x;
  _Float16* prow = pn_h + (size_t)c * FD + lane * 8;

  if (c >= NCLS) {                 // padding rows: all-zero
    v8h z = {};
    *(v8h*)prow = z;
    if (lane == 0) pn_sq[c] = 0.0f;
    return;
  }

  const float cnt = counts[c];
  const float* pr = protos + (size_t)c * FD + lane * 8;
  const float* sr = sums   + (size_t)c * FD + lane * 8;
  v4f p0 = *(const v4f*)pr, p1 = *(const v4f*)(pr + 4);
  v4f s0 = *(const v4f*)sr, s1 = *(const v4f*)(sr + 4);
  float p[8], mn[8];
  const float dc = 1.0f / fmaxf(cnt, 1.0f);
#pragma unroll
  for (int j = 0; j < 4; ++j) {
    p[j] = p0[j];  p[4 + j] = p1[j];
    mn[j] = s0[j] * dc;  mn[4 + j] = s1[j] * dc;
  }

  // class_mean = l2norm(sums / max(count,1))
  float ss = 0.0f;
#pragma unroll
  for (int j = 0; j < 8; ++j) ss += mn[j] * mn[j];
  ss = wave_sum(ss);
  const float im = 1.0f / fmaxf(sqrtf(ss), L2EPS);
#pragma unroll
  for (int j = 0; j < 8; ++j) mn[j] *= im;

  const float mom = (epoch_p[0] < 5) ? 0.0f : 0.99f;
  float cand[8];
  if (mom == 0.0f) {
#pragma unroll
    for (int j = 0; j < 8; ++j) cand[j] = mn[j];
  } else {
    float e[8];
#pragma unroll
    for (int j = 0; j < 8; ++j) e[j] = mom * p[j] + (1.0f - mom) * mn[j];
    float se = 0.0f;
#pragma unroll
    for (int j = 0; j < 8; ++j) se += e[j] * e[j];
    se = wave_sum(se);
    const float ie = 1.0f / fmaxf(sqrtf(se), L2EPS);
    const bool init = initialized[c] != 0;
#pragma unroll
    for (int j = 0; j < 8; ++j) cand[j] = init ? e[j] * ie : mn[j];
  }

  const bool present = cnt > 0.0f;
  float np[8];
#pragma unroll
  for (int j = 0; j < 8; ++j) np[j] = present ? cand[j] : p[j];

  // pn = l2norm(new_protos); store f16 + sum(pn*pn)
  float sn = 0.0f;
#pragma unroll
  for (int j = 0; j < 8; ++j) sn += np[j] * np[j];
  sn = wave_sum(sn);
  const float in2 = 1.0f / fmaxf(sqrtf(sn), L2EPS);

  v8h h;
  float s2 = 0.0f;
#pragma unroll
  for (int j = 0; j < 8; ++j) {
    float v = np[j] * in2;
    s2 += v * v;
    h[j] = (_Float16)v;
  }
  s2 = wave_sum(s2);
  *(v8h*)prow = h;
  if (lane == 0) pn_sq[c] = s2;
}

// ---------------------------------------------------------------------------
// Distance GEMM: out[m,c] = -sqrt(max(fsq[m] + pn_sq[c] - 2*dot(fn,pn), 0)).
// One wave owns a 16x64 output slab: one A fragment reused over 4 WMMA column
// tiles, K=256 in 8 steps of v_wmma_f32_16x16x32_f16.
//
// Fragment layouts used here (contiguous k order in A and B, not the ISA table order; wave32):
//   A (16x32 f16): lane L -> row m0+L%16; v16h elems 0..7  = K[k0+16*(L/16)..+7],
//                                          elems 8..15 = K[k0+16*(L/16)+8..+7]
//   B (32x16 f16): lane L -> col  c+L%16; v16h elems e = K[k0+16*(L/16)+e]
//                  (prototype rows are the B columns -> contiguous 32B load)
//   C/D (16x16 f32): acc[r], lane L -> row m0 + 8*(L/16) + r, col c + L%16
// ---------------------------------------------------------------------------
__global__ __launch_bounds__(256) void k_dist_wmma(
    const _Float16* __restrict__ fn_h, const _Float16* __restrict__ pn_h,
    const float* __restrict__ fsq, const float* __restrict__ pn_sq,
    float* __restrict__ out)
{
  const int tid  = blockIdx.x * blockDim.x + threadIdx.x;
  const int wave = tid >> 5;
  const int lane = tid & 31;
  const int cg   = wave & (CGROUPS - 1);   // column group (64 cols)
  const int mt   = wave >> 4;              // M tile
  const int half = lane >> 4;
  const int l16  = lane & 15;
  const int m0   = mt * 16;
  const int c0   = cg * 64;

  const _Float16* A  = fn_h + (size_t)(m0 + l16) * FD;
  const _Float16* B0 = pn_h + (size_t)(c0 +  0 + l16) * FD;
  const _Float16* B1 = pn_h + (size_t)(c0 + 16 + l16) * FD;
  const _Float16* B2 = pn_h + (size_t)(c0 + 32 + l16) * FD;
  const _Float16* B3 = pn_h + (size_t)(c0 + 48 + l16) * FD;

  v8f acc0 = {}, acc1 = {}, acc2 = {}, acc3 = {};

#pragma unroll
  for (int k0 = 0; k0 < FD; k0 += 32) {
    union { v16h v; v8h h[2]; } a;
    const int ka = k0 + half * 16;
    a.h[0] = *(const v8h*)(A + ka);
    a.h[1] = *(const v8h*)(A + ka + 8);

    const int kb = k0 + half * 16;
    v16h b0 = *(const v16h*)(B0 + kb);
    v16h b1 = *(const v16h*)(B1 + kb);
    v16h b2 = *(const v16h*)(B2 + kb);
    v16h b3 = *(const v16h*)(B3 + kb);

    acc0 = __builtin_amdgcn_wmma_f32_16x16x32_f16(false, a.v, false, b0,
                                                  (short)0, acc0, false, false);
    acc1 = __builtin_amdgcn_wmma_f32_16x16x32_f16(false, a.v, false, b1,
                                                  (short)0, acc1, false, false);
    acc2 = __builtin_amdgcn_wmma_f32_16x16x32_f16(false, a.v, false, b2,
                                                  (short)0, acc2, false, false);
    acc3 = __builtin_amdgcn_wmma_f32_16x16x32_f16(false, a.v, false, b3,
                                                  (short)0, acc3, false, false);
  }

  const int col0 = c0 +  0 + l16;
  const int col1 = c0 + 16 + l16;
  const int col2 = c0 + 32 + l16;
  const int col3 = c0 + 48 + l16;
  const float ps0 = pn_sq[col0];
  const float ps1 = pn_sq[col1];
  const float ps2 = pn_sq[col2];
  const float ps3 = pn_sq[col3];

#pragma unroll
  for (int r = 0; r < 8; ++r) {
    const int mrow = m0 + half * 8 + r;
    const float fs = fsq[mrow];
    float* orow = out + (size_t)mrow * NCLS;
    if (col0 < NCLS) orow[col0] = -sqrtf(fmaxf(fs + ps0 - 2.0f * acc0[r], 0.0f));
    if (col1 < NCLS) orow[col1] = -sqrtf(fmaxf(fs + ps1 - 2.0f * acc1[r], 0.0f));
    if (col2 < NCLS) orow[col2] = -sqrtf(fmaxf(fs + ps2 - 2.0f * acc2[r], 0.0f));
    if (col3 < NCLS) orow[col3] = -sqrtf(fmaxf(fs + ps3 - 2.0f * acc3[r], 0.0f));
  }
}

// ---------------------------------------------------------------------------
extern "C" void kernel_launch(void* const* d_in, const int* in_sizes, int n_in,
                              void* d_out, int out_size, void* d_ws, size_t ws_size,
                              hipStream_t stream) {
  (void)in_sizes; (void)n_in; (void)out_size; (void)ws_size;

  const float*         features    = (const float*)d_in[0];
  const float*         prototypes  = (const float*)d_in[1];
  const int*           labels      = (const int*)d_in[2];
  const unsigned char* initialized = (const unsigned char*)d_in[3]; // bool array
  const int*           epoch       = (const int*)d_in[4];
  float*               out         = (float*)d_out;

  // Workspace layout (~66 MB)
  char* ws = (char*)d_ws;
  size_t off = 0;
  _Float16* fn_h  = (_Float16*)(ws + off); off += (size_t)NROWS * FD * 2;
  _Float16* pn_h  = (_Float16*)(ws + off); off += (size_t)CPAD  * FD * 2;
  float*    fsq   = (float*)(ws + off);    off += (size_t)NROWS * 4;
  float*    pn_sq = (float*)(ws + off);    off += (size_t)CPAD  * 4;
  float*    sums  = (float*)(ws + off);    off += (size_t)NCLS * FD * 4;
  float*    counts= (float*)(ws + off);    off += (size_t)NCLS * 4;

  // 1) zero segment-sum accumulators (sums then counts, contiguous)
  {
    const int n = NCLS * FD + NCLS;
    k_zero<<<(n + 255) / 256, 256, 0, stream>>>(sums, n);
  }

  // 2) normalize features + segment sums (one wave per row)
  k_norm_accum<<<(NROWS * 32) / 256, 256, 0, stream>>>(
      features, labels, fn_h, fsq, sums, counts);

  // 3) prototype EMA update (one wave per padded class)
  k_proto<<<CPAD, 32, 0, stream>>>(
      prototypes, initialized, epoch, sums, counts, pn_h, pn_sq);

  // 4) WMMA distance GEMM: 8192 M-tiles x 16 column groups, 8 waves/block
  k_dist_wmma<<<(NROWS / 16) * CGROUPS * 32 / 256, 256, 0, stream>>>(
      fn_h, pn_h, fsq, pn_sq, out);
}
